// ComplexSympNet_61907658604593
// MI455X (gfx1250) — compile-verified, matrix operand order checked by reading, not run on hardware
//
#include <hip/hip_runtime.h>

// ---- problem constants (match reference) ----
#define NL    8
#define NDIM  128
#define LDIM  128
#define BROWS 65536

typedef __attribute__((ext_vector_type(16))) _Float16    v16h;
typedef __attribute__((ext_vector_type(8)))  _Float16    v8h;
typedef __attribute__((ext_vector_type(8)))  float       v8f;
typedef __attribute__((ext_vector_type(8)))  unsigned int v8u;

static __device__ __forceinline__ v8f wmma_f16(v16h a, v16h b, v8f c) {
  // D = A(16x32 f16) * B(32x16 f16) + C(16x16 f32)
  return __builtin_amdgcn_wmma_f32_16x16x32_f16(false, a, false, b, (short)0, c, false, false);
}

static __device__ __forceinline__ v16h neg16(v16h x) {
  union { v16h h; v8u u; } t;
  t.h = x;
  t.u = t.u ^ 0x80008000u;   // flip f16 sign bits (f16 WMMA has no A/B NEG modifier)
  return t.h;
}

// A-fragment (16x32 f16) from row-major LDS tile [16][136] halfs.
// lane<16: M=lane, K = kt*32 + {0..7, 16..23}; lane>=16: M=lane-16, K = kt*32 + {8..15, 24..31}
static __device__ __forceinline__ v16h loadA(const _Float16* base, int laneR, int laneH, int kt) {
  const _Float16* p = base + laneR * 136 + kt * 32 + laneH * 8;
  v8h lo = *(const v8h*)p;          // K group 0 (8 halfs, 16B aligned)
  v8h hi = *(const v8h*)(p + 16);   // K group 1
  return __builtin_shufflevector(lo, hi, 0,1,2,3,4,5,6,7,8,9,10,11,12,13,14,15);
}

// B-fragment from pre-swizzled workspace: fragment (ct,kt), 32 lanes x 16 halfs contiguous
static __device__ __forceinline__ v16h loadB(const _Float16* mat, int ct, int kt, int lane) {
  return *(const v16h*)(mat + (size_t)(((ct * 4 + kt) * 32 + lane) * 16));
}

// Hardware tanh on gfx1250 (v_tanh_f32 is a TRANS op that co-executes with VALU);
// fall back to the exp form if the builtin isn't declared by this toolchain.
static __device__ __forceinline__ float tanh_fast(float x) {
#if __has_builtin(__builtin_amdgcn_tanhf)
  return __builtin_amdgcn_tanhf(x);
#elif __has_builtin(__builtin_amdgcn_tanh_f32)
  return __builtin_amdgcn_tanh_f32(x);
#else
  float e = __expf(2.0f * x);
  return 1.0f - 2.0f / (e + 1.0f);   // saturates correctly at +-1
#endif
}

// ---------------------------------------------------------------------------
// Prep: build f16 WMMA-B-swizzled matrices per layer:
//   type 0: Wr^T   B[k=n][col=lo] = Wr[l][lo][n]
//   type 1: Wi^T
//   type 2: DWr    B[k=li][col=n] = diag[l][li]*Wr[l][li][n]
//   type 3: DWi
// Each matrix: 32 fragments * 32 lanes * 16 halfs = 32 KiB. Total 8*4*32KiB = 1 MiB.
// ---------------------------------------------------------------------------
__global__ void prep_weights_kernel(const float* __restrict__ Wr, const float* __restrict__ Wi,
                                    const float* __restrict__ diag, _Float16* __restrict__ wsh) {
  const int l    = blockIdx.x >> 2;
  const int type = blockIdx.x & 3;
  const int lane = threadIdx.x & 31;
  const int w    = threadIdx.x >> 5;      // 0..7
  const int laneR = lane & 15, laneH = lane >> 4;
  _Float16* out = wsh + (size_t)(l * 4 + type) * (32 * 32 * 16);

  for (int fo = 0; fo < 4; ++fo) {
    const int fi = fo * 8 + w;            // fragment index 0..31 = ct*4+kt
    const int ct = fi >> 2, kt = fi & 3;
    const int c  = ct * 16 + laneR;       // output column
    const int kb = kt * 32 + laneH * 8;   // K group base
    v16h vals;
#pragma unroll
    for (int e = 0; e < 16; ++e) {
      const int K = kb + (e < 8 ? e : 8 + e);  // e>=8 -> +16+(e-8)
      float v;
      if (type == 0)      v = Wr[((size_t)(l * LDIM) + c) * NDIM + K];
      else if (type == 1) v = Wi[((size_t)(l * LDIM) + c) * NDIM + K];
      else if (type == 2) v = diag[l * LDIM + K] * Wr[((size_t)(l * LDIM) + K) * NDIM + c];
      else                v = diag[l * LDIM + K] * Wi[((size_t)(l * LDIM) + K) * NDIM + c];
      vals[e] = (_Float16)v;
    }
    *(v16h*)(out + (size_t)((fi * 32 + lane) * 16)) = vals;
  }
}

// ---------------------------------------------------------------------------
// Fully fused SympNet: 4 waves/block, each wave owns 16 rows x 128 cols.
// State + accumulators live in VGPRs as WMMA C/D-layout tiles (v8f x 8 col-tiles).
// ---------------------------------------------------------------------------
__launch_bounds__(128, 1)
__global__ void symp_fused_kernel(const float* __restrict__ pr_in, const float* __restrict__ pi_in,
                                  const float* __restrict__ qr_in, const float* __restrict__ qi_in,
                                  const float* __restrict__ br,  const float* __restrict__ bi,
                                  const float* __restrict__ bias, const float* __restrict__ a,
                                  const _Float16* __restrict__ wsh,
                                  float* __restrict__ out) {
  __shared__ __align__(16) _Float16 buf[4][2][16][136];  // per-wave transpose staging (~34 KiB)

  const int lane  = threadIdx.x & 31;
  const int wave  = threadIdx.x >> 5;
  const int laneR = lane & 15;
  const int laneH = lane >> 4;
  const int rowBase = (blockIdx.x * 4 + wave) * 16;

  v8f qr[8], qi[8], pr[8], pi[8], qcr[8], qci[8], pcr[8], pci[8];

  // ---- load initial state in D-layout: element j -> row M = laneH*8+j, col = ct*16+laneR
#pragma unroll
  for (int ct = 0; ct < 8; ++ct) {
    const int col = ct * 16 + laneR;
#pragma unroll
    for (int j = 0; j < 8; ++j) {
      const size_t idx = (size_t)(rowBase + laneH * 8 + j) * NDIM + col;
      pr[ct][j] = pr_in[idx];
      pi[ct][j] = pi_in[idx];
      qr[ct][j] = qr_in[idx];
      qi[ct][j] = qi_in[idx];
    }
    qcr[ct] = qr[ct]; qci[ct] = qi[ct];
    pcr[ct] = pr[ct]; pci[ct] = pi[ct];
  }

  for (int l = 0; l < NL; ++l) {
    const float a_r = a[l * 4 + 0], a_i = a[l * 4 + 1];
    const float b_r = a[l * 4 + 2], b_i = a[l * 4 + 3];

    // ---- elementwise complex mix -> f16 -> LDS (row-major tiles)
#pragma unroll
    for (int ct = 0; ct < 8; ++ct) {
      const int col = ct * 16 + laneR;
#pragma unroll
      for (int j = 0; j < 8; ++j) {
        const int row = laneH * 8 + j;
        const float re = a_r * qr[ct][j] - a_i * qi[ct][j] + b_r * pr[ct][j] - b_i * pi[ct][j];
        const float im = a_r * qi[ct][j] + a_i * qr[ct][j] + b_r * pi[ct][j] + b_i * pr[ct][j];
        buf[wave][0][row][col] = (_Float16)re;
        buf[wave][1][row][col] = (_Float16)im;
      }
    }
    __syncthreads();

    v16h reA[4], imA[4], imNA[4];
#pragma unroll
    for (int kt = 0; kt < 4; ++kt) {
      reA[kt]  = loadA(&buf[wave][0][0][0], laneR, laneH, kt);
      imA[kt]  = loadA(&buf[wave][1][0][0], laneR, laneH, kt);
      imNA[kt] = neg16(imA[kt]);
    }
    __syncthreads();

    // ---- forward GEMMs: u = re@Wr^T - im@Wi^T (+br-bi); v = im@Wr^T + re@Wi^T (+br+bi)
    const _Float16* wrB = wsh + (size_t)(l * 4 + 0) * 16384;
    const _Float16* wiB = wsh + (size_t)(l * 4 + 1) * 16384;
#pragma unroll
    for (int ct = 0; ct < 8; ++ct) {
      v8f u = {}; v8f v = {};
#pragma unroll
      for (int kt = 0; kt < 4; ++kt) {
        const v16h wr = loadB(wrB, ct, kt, lane);
        const v16h wi = loadB(wiB, ct, kt, lane);
        u = wmma_f16(reA[kt],  wr, u);
        u = wmma_f16(imNA[kt], wi, u);
        v = wmma_f16(imA[kt],  wr, v);
        v = wmma_f16(reA[kt],  wi, v);
      }
      const int col = ct * 16 + laneR;
      const float brv = br[l * LDIM + col], biv = bi[l * LDIM + col];
#pragma unroll
      for (int j = 0; j < 8; ++j) {
        const int row = laneH * 8 + j;
        buf[wave][0][row][col] = (_Float16)tanh_fast(u[j] + (brv - biv));
        buf[wave][1][row][col] = (_Float16)tanh_fast(v[j] + (brv + biv));
      }
    }
    __syncthreads();

#pragma unroll
    for (int kt = 0; kt < 4; ++kt) {
      reA[kt]  = loadA(&buf[wave][0][0][0], laneR, laneH, kt);
      imA[kt]  = loadA(&buf[wave][1][0][0], laneR, laneH, kt);
      imNA[kt] = neg16(imA[kt]);
    }

    // ---- backward GEMMs: r2 = re_@DWr - im_@DWi ; i2 = im_@DWr + re_@DWi + bias
    const _Float16* dwrB = wsh + (size_t)(l * 4 + 2) * 16384;
    const _Float16* dwiB = wsh + (size_t)(l * 4 + 3) * 16384;
#pragma unroll
    for (int ct = 0; ct < 8; ++ct) {
      v8f r2 = {}; v8f i2 = {};
#pragma unroll
      for (int kt = 0; kt < 4; ++kt) {
        const v16h dr = loadB(dwrB, ct, kt, lane);
        const v16h di = loadB(dwiB, ct, kt, lane);
        r2 = wmma_f16(reA[kt],  dr, r2);
        r2 = wmma_f16(imNA[kt], di, r2);
        i2 = wmma_f16(imA[kt],  dr, i2);
        i2 = wmma_f16(reA[kt],  di, i2);
      }
      const int col = ct * 16 + laneR;
      const float bsv = bias[l * NDIM + col];
#pragma unroll
      for (int j = 0; j < 8; ++j) {
        const float R = r2[j];
        const float I = i2[j] + bsv;
        qr[ct][j] += b_r * R - b_i * I;
        qi[ct][j] += b_r * I + b_i * R;
        pr[ct][j] -= a_r * R - a_i * I;
        pi[ct][j] -= a_r * I + a_i * R;
        qcr[ct][j] += qr[ct][j];
        qci[ct][j] += qi[ct][j];
        pcr[ct][j] += pr[ct][j];
        pci[ct][j] += pi[ct][j];
      }
    }
    __syncthreads();
  }

  // ---- store outputs: stack([pc_r, pc_i, qc_r, qc_i]) -> [4][B][N]
  const size_t plane = (size_t)BROWS * NDIM;
#pragma unroll
  for (int ct = 0; ct < 8; ++ct) {
    const int col = ct * 16 + laneR;
#pragma unroll
    for (int j = 0; j < 8; ++j) {
      const size_t idx = (size_t)(rowBase + laneH * 8 + j) * NDIM + col;
      out[idx]             = pcr[ct][j];
      out[plane + idx]     = pci[ct][j];
      out[2 * plane + idx] = qcr[ct][j];
      out[3 * plane + idx] = qci[ct][j];
    }
  }
}

extern "C" void kernel_launch(void* const* d_in, const int* in_sizes, int n_in,
                              void* d_out, int out_size, void* d_ws, size_t ws_size,
                              hipStream_t stream) {
  const float* p_r  = (const float*)d_in[0];
  const float* p_i  = (const float*)d_in[1];
  const float* q_r  = (const float*)d_in[2];
  const float* q_i  = (const float*)d_in[3];
  const float* Wr   = (const float*)d_in[4];
  const float* Wi   = (const float*)d_in[5];
  const float* br   = (const float*)d_in[6];
  const float* bi   = (const float*)d_in[7];
  const float* bias = (const float*)d_in[8];
  const float* diag = (const float*)d_in[9];
  const float* a    = (const float*)d_in[10];

  _Float16* wsh = (_Float16*)d_ws;  // needs 1 MiB of scratch for swizzled f16 weights

  // 1) swizzle weights into WMMA-B f16 fragment layout (stream-ordered before main)
  prep_weights_kernel<<<NL * 4, 256, 0, stream>>>(Wr, Wi, diag, wsh);

  // 2) fully fused 8-layer SympNet: 1024 blocks x 128 threads (4 wave32 each, 64 rows/block)
  symp_fused_kernel<<<BROWS / 64, 128, 0, stream>>>(p_r, p_i, q_r, q_i, br, bi, bias, a,
                                                    wsh, (float*)d_out);
}
